// AutoBoost_21285857919350
// MI455X (gfx1250) — matrix operand order repaired and checked by reading, not compiled, not run on hardware
//
#include <hip/hip_runtime.h>
#include <hip/hip_bf16.h>
#include <math.h>

// ---------------- problem constants ----------------
#define Bn    48
#define TLEN  64000
#define NFk   5
#define NBk   5
#define Ck    101        // section length
#define LFILT 501        // composite filter length
#define PI_F  3.14159265358979f
#define MINF_  20.0f
#define MAXF_  7980.0f   // FS/2 - 20
#define MINBW_ 100.0f
#define MAXBW_ 1000.0f

// FIR tiling: one 256-thread block = 8 waves = 8 tiles of 256 outputs
#define NTILES    250            // TLEN / 256
#define BLK_OUT   2048           // 8 * 256
#define XSTAGE    2576           // staged samples per power version (max index 2575)
#define FPAD      560            // 16 lead-in + 544 band columns

typedef __attribute__((ext_vector_type(16))) _Float16 v16h;
typedef __attribute__((ext_vector_type(8)))  float    v8f;

// ---------------- stateless RNG (threefry stand-in) ----------------
__device__ __forceinline__ unsigned int hash32(unsigned int a, unsigned int b, unsigned int c) {
    unsigned int h = a * 0x9E3779B1u ^ b * 0x85EBCA77u ^ c * 0xC2B2AE3Du;
    h ^= h >> 16; h *= 0x7FEB352Du;
    h ^= h >> 15; h *= 0x846CA68Bu;
    h ^= h >> 16;
    return h;
}
__device__ __forceinline__ float u01(unsigned int h) {
    return (float)(h >> 8) * (1.0f / 16777216.0f);
}
__device__ __forceinline__ float gaussf(unsigned int h1, unsigned int h2) {
    float u1 = fmaxf(u01(h1), 1e-7f);
    float u2 = u01(h2);
    return sqrtf(-2.0f * logf(u1)) * cosf(2.0f * PI_F * u2);
}
__device__ __forceinline__ float sincf(float x) {
    float ax = fabsf(x);
    if (ax < 1e-6f) return 1.0f;
    float px = PI_F * x;
    return sinf(px) / px;
}
__device__ __forceinline__ float sigmoidf(float x) { return 1.0f / (1.0f + expf(-x)); }

// ---------------- generic conv1d + BN + ReLU (writes h1 buffer) ----------------
__global__ void conv1d_bn_relu(const float* __restrict__ in, int Cin, int Tin,
                               const float* __restrict__ w, const float* __restrict__ bias,
                               const float* __restrict__ g, const float* __restrict__ bb,
                               float* __restrict__ out, int h1_stride,
                               int Cout, int Tout, int K, int S, int Pd)
{
    int b  = blockIdx.z;
    int co = blockIdx.y;
    int t  = blockIdx.x * blockDim.x + threadIdx.x;
    if (t >= Tout) return;
    float acc = bias[co];
    const float* wp = w + (size_t)co * Cin * K;
    int base = t * S - Pd;
    for (int ci = 0; ci < Cin; ++ci) {
        const float* ip = in + ((size_t)b * Cin + ci) * Tin;
        for (int k = 0; k < K; ++k) {
            int tt = base + k;
            float v = (tt >= 0 && tt < Tin) ? ip[tt] : 0.0f;
            acc += v * wp[ci * K + k];
        }
    }
    float gs = g[co] * rsqrtf(1.0f + 1e-5f);
    acc = acc * gs + bb[co];
    out[(size_t)b * h1_stride + (size_t)co * Tout + t] = fmaxf(acc, 0.0f);
}

// ---------------- conv2 + BN + ReLU fused with temporal mean ----------------
__global__ void conv2_bn_relu_mean(const float* __restrict__ h1, int h1_stride, int Cin, int Tin,
                                   const float* __restrict__ w, const float* __restrict__ bias,
                                   const float* __restrict__ g, const float* __restrict__ bb,
                                   float* __restrict__ feat, int Cout, int Tout, int K, int S, int Pd)
{
    int co = blockIdx.x;
    int b  = blockIdx.y;
    float gs = g[co] * rsqrtf(1.0f + 1e-5f);
    float bco = bias[co], sco = bb[co];
    float sum = 0.0f;
    for (int t = threadIdx.x; t < Tout; t += blockDim.x) {
        float acc = bco;
        int base = t * S - Pd;
        for (int ci = 0; ci < Cin; ++ci) {
            const float* ip = h1 + (size_t)b * h1_stride + (size_t)ci * Tin;
            const float* wp = w + ((size_t)co * Cin + ci) * K;
            for (int k = 0; k < K; ++k) {
                int tt = base + k;
                float v = (tt >= 0 && tt < Tin) ? ip[tt] : 0.0f;
                acc += v * wp[k];
            }
        }
        sum += fmaxf(acc * gs + sco, 0.0f);
    }
    __shared__ float red[256];
    red[threadIdx.x] = sum;
    __syncthreads();
    for (int s = blockDim.x >> 1; s > 0; s >>= 1) {
        if (threadIdx.x < (unsigned)s) red[threadIdx.x] += red[threadIdx.x + s];
        __syncthreads();
    }
    if (threadIdx.x == 0) feat[(size_t)b * Cout + co] = red[0] / (float)Tout;
}

// ---------------- FC + head(s); mode 0=sigmoid, 1=sigmoid+clip, 2=softmax ----------------
__global__ void head_kernel(const float* __restrict__ feat, int fdim,
                            const float* __restrict__ fcw, const float* __restrict__ fcb, int hdim,
                            const float* __restrict__ h1w, const float* __restrict__ h1b, int o1, float* __restrict__ out1,
                            const float* __restrict__ h2w, const float* __restrict__ h2b, int o2, float* __restrict__ out2,
                            int mode)
{
    int b = blockIdx.x;
    __shared__ float fl[16];
    __shared__ float hl[32];
    __shared__ float ol[32];
    int tid = threadIdx.x;
    if (tid < fdim) fl[tid] = feat[(size_t)b * fdim + tid];
    __syncthreads();
    for (int j = tid; j < hdim; j += blockDim.x) {
        float a = fcb[j];
        for (int i = 0; i < fdim; ++i) a += fcw[j * fdim + i] * fl[i];
        hl[j] = fmaxf(a, 0.0f);
    }
    __syncthreads();
    for (int j = tid; j < o1; j += blockDim.x) {
        float a = h1b[j];
        for (int i = 0; i < hdim; ++i) a += h1w[j * hdim + i] * hl[i];
        if (mode == 2) {
            ol[j] = a;
        } else {
            float s = sigmoidf(a);
            if (mode == 1) s = fminf(fmaxf(s, 0.05f), 1.0f);
            out1[(size_t)b * o1 + j] = s;
        }
    }
    if (mode == 2) {
        __syncthreads();
        if (tid == 0) {
            float mx = -1e30f;
            for (int j = 0; j < o1; ++j) mx = fmaxf(mx, ol[j]);
            float s = 0.0f;
            for (int j = 0; j < o1; ++j) { ol[j] = expf(ol[j] - mx); s += ol[j]; }
            for (int j = 0; j < o1; ++j) out1[(size_t)b * o1 + j] = ol[j] / s;
        }
    }
    if (o2 > 0) {
        for (int j = tid; j < o2; j += blockDim.x) {
            float a = h2b[j];
            for (int i = 0; i < hdim; ++i) a += h2w[j * hdim + i] * hl[i];
            out2[(size_t)b * o2 + j] = sigmoidf(a);
        }
    }
}

// ---------------- gen_notch: build composite 501-tap filter, reversed + gain-normalized ----------------
__global__ void gen_notch_kernel(const float* __restrict__ lnl_fc_u, const float* __restrict__ lnl_bw_u,
                                 const float* __restrict__ ssi_fc_u, const float* __restrict__ ssi_bw_u,
                                 float* __restrict__ filt /* Bn*6*512 */)
{
    int f = blockIdx.x;   // 0..4 = LNL filters, 5 = SSI
    int b = blockIdx.y;
    __shared__ float hbs[NBk][Ck];
    __shared__ float buf0[544], buf1[544];
    __shared__ float redmax[128];

    float gain_db = (f < 5) ? (-10.0f + 10.0f * u01(hash32(0u, b, f)))
                            : (-10.0f + 10.0f * u01(hash32(1u, b, 0u)));

    for (int idx = threadIdx.x; idx < NBk * Ck; idx += blockDim.x) {
        int s = idx / Ck, c = idx % Ck;
        float fcu = (f < 5) ? lnl_fc_u[(size_t)b * 25 + f * 5 + s] : ssi_fc_u[(size_t)b * 5 + s];
        float bwu = (f < 5) ? lnl_bw_u[(size_t)b * 25 + f * 5 + s] : ssi_bw_u[(size_t)b * 5 + s];
        float fc = MINF_ + fcu * (MAXF_ - MINF_);
        float bw = MINBW_ + bwu * (MAXBW_ - MINBW_);
        float nyq = 8000.0f;
        float f1 = fmaxf(fc - 0.5f * bw, 0.001f) / nyq;
        float f2 = fminf(fc + 0.5f * bw, nyq - 0.001f) / nyq;
        float m  = (float)c - 50.0f;
        float hbp = f2 * sincf(f2 * m) - f1 * sincf(f1 * m);
        float delta = (c == 50) ? 1.0f : 0.0f;
        float wnd = 0.54f - 0.46f * cosf(2.0f * PI_F * (float)c / 100.0f);
        hbs[s][c] = (delta - hbp) * wnd;
    }
    __syncthreads();
    for (int i = threadIdx.x; i < 544; i += blockDim.x) buf0[i] = (i < Ck) ? hbs[0][i] : 0.0f;
    __syncthreads();

    int len = Ck;
    float* cur = buf0;
    float* nxt = buf1;
    for (int s = 1; s < NBk; ++s) {
        int nl = len + Ck - 1;
        for (int n = threadIdx.x; n < 544; n += blockDim.x) {
            float acc = 0.0f;
            if (n < nl) {
                int k0 = n - (Ck - 1); if (k0 < 0) k0 = 0;
                int k1 = (n < len - 1) ? n : (len - 1);
                for (int k = k0; k <= k1; ++k) acc += cur[k] * hbs[s][n - k];
            }
            nxt[n] = acc;
        }
        __syncthreads();
        float* tmp = cur; cur = nxt; nxt = tmp;
        len = nl;
    }

    // max |rfft(b, n=1024)| via direct DFT over 513 bins
    float lmax = 0.0f;
    for (int bin = threadIdx.x; bin < 513; bin += blockDim.x) {
        float re = 0.0f, im = 0.0f;
        float w0 = -2.0f * PI_F * (float)bin / 1024.0f;
        for (int n = 0; n < LFILT; ++n) {
            float a = w0 * (float)n;
            re += cur[n] * cosf(a);
            im += cur[n] * sinf(a);
        }
        lmax = fmaxf(lmax, sqrtf(re * re + im * im));
    }
    redmax[threadIdx.x] = lmax;
    __syncthreads();
    for (int s = 64; s > 0; s >>= 1) {
        if ((int)threadIdx.x < s) redmax[threadIdx.x] = fmaxf(redmax[threadIdx.x], redmax[threadIdx.x + s]);
        __syncthreads();
    }
    float scale = powf(10.0f, gain_db / 20.0f) / fmaxf(redmax[0], 1e-8f);
    float* out = filt + ((size_t)b * 6 + f) * 512;
    for (int j = threadIdx.x; j < 512; j += blockDim.x)
        out[j] = (j <= 500) ? cur[500 - j] * scale : 0.0f;   // store reversed
}

// ---------------- WMMA banded-Toeplitz FIR (the compute core) ----------------
// Block = 256 threads (8 waves), covers 8 tiles of 256 outputs for one (path, batch).
// LDS staging (all f16): padded filter band arrays + per-power signal segments.
// Inner loop per chunk: 2x ds_load_b128 (A band) + 2x ds_load_b128 (B, one aligned
// v16h load) + v_wmma -- no converts, no branches.
template<int NFILT>
__device__ __forceinline__ void fir_wave_compute(const _Float16* __restrict__ fsh,
                                                 const _Float16* __restrict__ xph,
                                                 int lt0, int hi, int lo,
                                                 v8f& total)
{
    #pragma unroll
    for (int fl = 0; fl < NFILT; ++fl) {
        const _Float16* fp = fsh + fl * FPAD;
        const _Float16* xp = xph + fl * XSTAGE;
        v8f acc = {0.f,0.f,0.f,0.f,0.f,0.f,0.f,0.f};
        for (int ch = 0; ch < 17; ++ch) {
            const int kc = ch * 32;
            // A tile (16x32 f16): A[m][c] = band[16 + c - m]; zero-padded in LDS
            v16h Am;
            const int abase = 16 + kc + (hi << 4) - lo;
            #pragma unroll
            for (int e = 0; e < 16; ++e) {
                int kl = e;        // element -> K offset (contiguous: k = 16*hi + e)
                Am[e] = fp[abase + kl];
            }
            // B tile (32x16 f16): 16 consecutive staged halves, 32B-aligned
            const int jbase = lt0 + 16 * lo + kc + (hi << 4);
            v16h Bm = *reinterpret_cast<const v16h*>(xp + jbase);
            acc = __builtin_amdgcn_wmma_f32_16x16x32_f16(false, Am, false, Bm,
                                                         (short)0, acc, false, false);
        }
        total = total + acc;
    }
}

__global__ __launch_bounds__(256) void fir_wmma_kernel(const float* __restrict__ x,
                                                       const float* __restrict__ filt,
                                                       float* __restrict__ inp)
{
    __shared__ _Float16 fsh[5 * FPAD];                       // up to 5 padded band filters
    __shared__ __align__(32) _Float16 xph[5 * XSTAGE];       // x^1..x^5 (LNL) or noise (SSI)

    const int blk  = blockIdx.x;         // 32 blocks of 8 tiles
    const int path = blockIdx.y;         // 0 = LNL, 1 = SSI
    const int b    = blockIdx.z;
    const int tid  = threadIdx.x;
    const int bt0  = blk * BLK_OUT;      // first output sample of this block

    // ---- stage filters (zero-padded band arrays, f16) ----
    const int nf = (path == 0) ? 5 : 1;
    for (int j = tid; j < nf * FPAD; j += 256) {
        int fl  = j / FPAD;
        int jj  = j - fl * FPAD;
        int f   = (path == 0) ? fl : 5;
        int idx = jj - 16;
        float v = (idx >= 0 && idx <= 500) ? filt[((size_t)b * 6 + f) * 512 + idx] : 0.0f;
        fsh[j] = (_Float16)v;
    }
    // ---- stage signal segment [bt0-250, bt0+2325] with boundary zeros ----
    const float* xb = x + (size_t)b * TLEN;
    for (int j = tid; j < XSTAGE; j += 256) {
        int t = bt0 - 250 + j;
        bool ok = (t >= 0 && t < TLEN);
        if (path == 0) {
            float xv = ok ? xb[t] : 0.0f;
            float v = xv;
            xph[0 * XSTAGE + j] = (_Float16)v;
            v *= xv; xph[1 * XSTAGE + j] = (_Float16)v;
            v *= xv; xph[2 * XSTAGE + j] = (_Float16)v;
            v *= xv; xph[3 * XSTAGE + j] = (_Float16)v;
            v *= xv; xph[4 * XSTAGE + j] = (_Float16)v;
        } else {
            float v = ok ? gaussf(hash32(6u, (unsigned)b, (unsigned)t),
                                  hash32(7u, (unsigned)b, (unsigned)t))
                         : 0.0f;
            xph[j] = (_Float16)v;
        }
    }
    __syncthreads();

    // ---- per-wave WMMA FIR over one 256-output tile ----
    const int wv   = tid >> 5;
    const int lane = tid & 31;
    const int hi   = lane >> 4;
    const int lo   = lane & 15;
    int tile = blk * 8 + wv;
    if (tile > NTILES - 1) tile = NTILES - 1;     // clamp (duplicate write of same data)
    const int lt0 = (tile - blk * 8) * 256;       // wave's offset inside staged segment

    v8f total = {0.f,0.f,0.f,0.f,0.f,0.f,0.f,0.f};
    if (path == 0) fir_wave_compute<5>(fsh, xph, lt0, hi, lo, total);
    else           fir_wave_compute<1>(fsh, xph, lt0, hi, lo, total);

    // D layout: element r -> M = r + 8*hi, N = lo; t = tile*256 + M + 16N
    const int outch = (path == 0) ? 1 : 3;
    float* dst = inp + ((size_t)b * 4 + outch) * TLEN + tile * 256 + hi * 8 + lo * 16;
    #pragma unroll
    for (int r = 0; r < 8; ++r) dst[r] = total[r];
}

// ---------------- per-batch post-processing ----------------
__global__ void copy_x_kernel(const float* __restrict__ x, float* __restrict__ inp)
{
    int b = blockIdx.y;
    int t = blockIdx.x * blockDim.x + threadIdx.x;
    if (t >= TLEN) return;
    inp[(size_t)b * 4 * TLEN + t] = x[(size_t)b * TLEN + t];
}

__global__ void lnl_post_kernel(float* __restrict__ inp)
{
    int b = blockIdx.x;
    float* y = inp + ((size_t)b * 4 + 1) * TLEN;
    __shared__ float red[256];
    float s = 0.0f;
    for (int t = threadIdx.x; t < TLEN; t += 256) s += y[t];
    red[threadIdx.x] = s;
    __syncthreads();
    for (int k = 128; k > 0; k >>= 1) { if ((int)threadIdx.x < k) red[threadIdx.x] += red[threadIdx.x + k]; __syncthreads(); }
    float mean = red[0] / (float)TLEN;
    __syncthreads();
    float mx = 0.0f;
    for (int t = threadIdx.x; t < TLEN; t += 256) {
        float v = y[t] - mean;
        y[t] = v;
        mx = fmaxf(mx, fabsf(v));
    }
    red[threadIdx.x] = mx;
    __syncthreads();
    for (int k = 128; k > 0; k >>= 1) { if ((int)threadIdx.x < k) red[threadIdx.x] = fmaxf(red[threadIdx.x], red[threadIdx.x + k]); __syncthreads(); }
    float m = red[0];
    float scale = (m > 1.0f) ? 1.0f / (m + 1e-8f) : 1.0f;
    for (int t = threadIdx.x; t < TLEN; t += 256) y[t] *= scale;
}

__global__ void isd_post_kernel(const float* __restrict__ x, const float* __restrict__ density,
                                float* __restrict__ inp)
{
    int b = blockIdx.x;
    float* y = inp + ((size_t)b * 4 + 2) * TLEN;
    const float* xb = x + (size_t)b * TLEN;
    float dens = density[b];
    __shared__ float red[256];
    float mx = 0.0f;
    for (int t = threadIdx.x; t < TLEN; t += 256) {
        float xv = xb[t];
        float u  = u01(hash32(3u, b, t));
        float fr = (2.0f * u01(hash32(4u, b, t)) - 1.0f) * (2.0f * u01(hash32(5u, b, t)) - 1.0f);
        float mask = sigmoidf((dens - u) * 10.0f);
        float v = xv + mask * (2.0f * xv * fr);   // gain_isd = 2.0
        y[t] = v;
        mx = fmaxf(mx, fabsf(v));
    }
    red[threadIdx.x] = mx;
    __syncthreads();
    for (int k = 128; k > 0; k >>= 1) { if ((int)threadIdx.x < k) red[threadIdx.x] = fmaxf(red[threadIdx.x], red[threadIdx.x + k]); __syncthreads(); }
    float m = red[0];
    float scale = (m > 1.0f) ? 1.0f / (m + 1e-8f) : 1.0f;
    for (int t = threadIdx.x; t < TLEN; t += 256) y[t] *= scale;
}

__global__ void ssi_post_kernel(const float* __restrict__ x, float* __restrict__ inp)
{
    int b = blockIdx.x;
    float* nc = inp + ((size_t)b * 4 + 3) * TLEN;
    const float* xb = x + (size_t)b * TLEN;
    __shared__ float r1[256], r2[256], r3[256];
    float mx = 0.0f, sn = 0.0f, sx = 0.0f;
    for (int t = threadIdx.x; t < TLEN; t += 256) {
        float v = nc[t];
        mx = fmaxf(mx, fabsf(v));
        sn += v * v;
        float xv = xb[t];
        sx += xv * xv;
    }
    r1[threadIdx.x] = mx; r2[threadIdx.x] = sn; r3[threadIdx.x] = sx;
    __syncthreads();
    for (int k = 128; k > 0; k >>= 1) {
        if ((int)threadIdx.x < k) {
            r1[threadIdx.x] = fmaxf(r1[threadIdx.x], r1[threadIdx.x + k]);
            r2[threadIdx.x] += r2[threadIdx.x + k];
            r3[threadIdx.x] += r3[threadIdx.x + k];
        }
        __syncthreads();
    }
    float m  = r1[0] + 1e-8f;
    float ne = sqrtf(r2[0]) / m;
    float xe = sqrtf(r3[0]);
    float snr = 10.0f + 30.0f * u01(hash32(2u, b, 0u));
    float target = xe / (powf(10.0f, snr / 20.0f) + 1e-8f);
    float s = (target / (ne + 1e-8f)) / m;
    __syncthreads();
    for (int t = threadIdx.x; t < TLEN; t += 256) nc[t] = xb[t] + nc[t] * s;
}

__global__ void mix_kernel(const float* __restrict__ inp, const float* __restrict__ w,
                           float* __restrict__ out)
{
    int b = blockIdx.y;
    int t = blockIdx.x * blockDim.x + threadIdx.x;
    if (t >= TLEN) return;
    const float* wb = w + (size_t)b * 3;
    float acc = 0.0f;
    for (int c = 0; c < 3; ++c) acc += wb[c] * inp[((size_t)b * 4 + 1 + c) * TLEN + t];
    out[(size_t)b * TLEN + t] = acc;
}

// ---------------- parameter indices: sorted(PARAM_SHAPES), d_in[1+i] ----------------
enum {
    ISD_BN1_B=0, ISD_BN1_G, ISD_BN2_B, ISD_BN2_G, ISD_CONV1_B, ISD_CONV1_W, ISD_CONV2_B, ISD_CONV2_W,
    ISD_FC_B, ISD_FC_W, ISD_HEAD_B, ISD_HEAD_W,
    LNL_BN1_B, LNL_BN1_G, LNL_BN2_B, LNL_BN2_G, LNL_CONV1_B, LNL_CONV1_W, LNL_CONV2_B, LNL_CONV2_W,
    LNL_FC_B, LNL_FC_W, LNL_HBW_B, LNL_HBW_W, LNL_HFC_B, LNL_HFC_W,
    RT_BN1_B, RT_BN1_G, RT_BN2_B, RT_BN2_G, RT_CONV1_B, RT_CONV1_W, RT_CONV2_B, RT_CONV2_W,
    RT_FC_B, RT_FC_W, RT_HEAD_B, RT_HEAD_W,
    SSI_BN1_B, SSI_BN1_G, SSI_BN2_B, SSI_BN2_G, SSI_CONV1_B, SSI_CONV1_W, SSI_CONV2_B, SSI_CONV2_W,
    SSI_FC_B, SSI_FC_W, SSI_HBW_B, SSI_HBW_W, SSI_HFC_B, SSI_HFC_W
};

extern "C" void kernel_launch(void* const* d_in, const int* in_sizes, int n_in,
                              void* d_out, int out_size, void* d_ws, size_t ws_size,
                              hipStream_t stream)
{
    (void)in_sizes; (void)n_in; (void)out_size; (void)ws_size;
    const float* x = (const float*)d_in[0];
    auto P = [&](int i) { return (const float*)d_in[1 + i]; };

    // -------- workspace layout (floats) --------
    const size_t H1S      = (size_t)8 * 16008;                  // per-batch h1 stride (max)
    float* ws = (float*)d_ws;
    float* inp       = ws;                                      // Bn*4*TLEN (ch0=x, ch1=lnl, ch2=isd, ch3=ssi)
    float* h1        = inp + (size_t)Bn * 4 * TLEN;             // Bn*H1S
    float* filt      = h1  + (size_t)Bn * H1S;                  // Bn*6*512 (reversed, gain-scaled)
    float* feat_lnl  = filt + (size_t)Bn * 6 * 512;             // Bn*16
    float* feat_isd  = feat_lnl + (size_t)Bn * 16;              // Bn*8 (stride 8)
    float* feat_ssi  = feat_isd + (size_t)Bn * 16;              // Bn*16
    float* feat_rt   = feat_ssi + (size_t)Bn * 16;              // Bn*16
    float* lnl_fc_u  = feat_rt  + (size_t)Bn * 16;              // Bn*25
    float* lnl_bw_u  = lnl_fc_u + (size_t)Bn * 25;              // Bn*25
    float* density   = lnl_bw_u + (size_t)Bn * 25;              // Bn
    float* ssi_fc_u  = density  + (size_t)Bn;                   // Bn*5
    float* ssi_bw_u  = ssi_fc_u + (size_t)Bn * 5;               // Bn*5
    float* rt_w      = ssi_bw_u + (size_t)Bn * 5;               // Bn*3

    dim3 blk(256);

    // -------- LNL encoder: conv1(8,64,s4,p32) -> conv2(16,32,s4,p16)+mean -> heads --------
    conv1d_bn_relu<<<dim3((16001 + 255) / 256, 8, Bn), blk, 0, stream>>>(
        x, 1, TLEN, P(LNL_CONV1_W), P(LNL_CONV1_B), P(LNL_BN1_G), P(LNL_BN1_B),
        h1, (int)H1S, 8, 16001, 64, 4, 32);
    conv2_bn_relu_mean<<<dim3(16, Bn), blk, 0, stream>>>(
        h1, (int)H1S, 8, 16001, P(LNL_CONV2_W), P(LNL_CONV2_B), P(LNL_BN2_G), P(LNL_BN2_B),
        feat_lnl, 16, 4001, 32, 4, 16);
    head_kernel<<<Bn, 32, 0, stream>>>(feat_lnl, 16, P(LNL_FC_W), P(LNL_FC_B), 32,
        P(LNL_HFC_W), P(LNL_HFC_B), 25, lnl_fc_u,
        P(LNL_HBW_W), P(LNL_HBW_B), 25, lnl_bw_u, 0);

    // -------- ISD encoder: conv1(4,32,s4,p16) -> conv2(8,16,s4,p8)+mean -> head --------
    conv1d_bn_relu<<<dim3((16001 + 255) / 256, 4, Bn), blk, 0, stream>>>(
        x, 1, TLEN, P(ISD_CONV1_W), P(ISD_CONV1_B), P(ISD_BN1_G), P(ISD_BN1_B),
        h1, (int)H1S, 4, 16001, 32, 4, 16);
    conv2_bn_relu_mean<<<dim3(8, Bn), blk, 0, stream>>>(
        h1, (int)H1S, 4, 16001, P(ISD_CONV2_W), P(ISD_CONV2_B), P(ISD_BN2_G), P(ISD_BN2_B),
        feat_isd, 8, 4001, 16, 4, 8);
    head_kernel<<<Bn, 32, 0, stream>>>(feat_isd, 8, P(ISD_FC_W), P(ISD_FC_B), 16,
        P(ISD_HEAD_W), P(ISD_HEAD_B), 1, density,
        nullptr, nullptr, 0, nullptr, 1);

    // -------- SSI encoder --------
    conv1d_bn_relu<<<dim3((16001 + 255) / 256, 8, Bn), blk, 0, stream>>>(
        x, 1, TLEN, P(SSI_CONV1_W), P(SSI_CONV1_B), P(SSI_BN1_G), P(SSI_BN1_B),
        h1, (int)H1S, 8, 16001, 64, 4, 32);
    conv2_bn_relu_mean<<<dim3(16, Bn), blk, 0, stream>>>(
        h1, (int)H1S, 8, 16001, P(SSI_CONV2_W), P(SSI_CONV2_B), P(SSI_BN2_G), P(SSI_BN2_B),
        feat_ssi, 16, 4001, 32, 4, 16);
    head_kernel<<<Bn, 32, 0, stream>>>(feat_ssi, 16, P(SSI_FC_W), P(SSI_FC_B), 32,
        P(SSI_HFC_W), P(SSI_HFC_B), 5, ssi_fc_u,
        P(SSI_HBW_W), P(SSI_HBW_B), 5, ssi_bw_u, 0);

    // -------- synthesize all 6 composite filters per batch --------
    gen_notch_kernel<<<dim3(6, Bn), 128, 0, stream>>>(lnl_fc_u, lnl_bw_u, ssi_fc_u, ssi_bw_u, filt);

    // -------- augmentation: x copy + WMMA FIRs + per-batch normalizations --------
    copy_x_kernel<<<dim3((TLEN + 255) / 256, Bn), blk, 0, stream>>>(x, inp);
    fir_wmma_kernel<<<dim3((NTILES + 7) / 8, 2, Bn), 256, 0, stream>>>(x, filt, inp);
    lnl_post_kernel<<<Bn, 256, 0, stream>>>(inp);
    isd_post_kernel<<<Bn, 256, 0, stream>>>(x, density, inp);
    ssi_post_kernel<<<Bn, 256, 0, stream>>>(x, inp);

    // -------- router: conv1(8,4,128,s8,p64) -> conv2(16,64,s8,p32)+mean -> softmax head --------
    conv1d_bn_relu<<<dim3((8001 + 255) / 256, 8, Bn), blk, 0, stream>>>(
        inp, 4, TLEN, P(RT_CONV1_W), P(RT_CONV1_B), P(RT_BN1_G), P(RT_BN1_B),
        h1, (int)H1S, 8, 8001, 128, 8, 64);
    conv2_bn_relu_mean<<<dim3(16, Bn), blk, 0, stream>>>(
        h1, (int)H1S, 8, 8001, P(RT_CONV2_W), P(RT_CONV2_B), P(RT_BN2_G), P(RT_BN2_B),
        feat_rt, 16, 1001, 64, 8, 32);
    head_kernel<<<Bn, 32, 0, stream>>>(feat_rt, 16, P(RT_FC_W), P(RT_FC_B), 32,
        P(RT_HEAD_W), P(RT_HEAD_B), 3, rt_w,
        nullptr, nullptr, 0, nullptr, 2);

    // -------- weighted mix --------
    mix_kernel<<<dim3((TLEN + 255) / 256, Bn), blk, 0, stream>>>(inp, rt_w, (float*)d_out);
}
